// DINOVae_15753940042111
// MI455X (gfx1250) — matrix operand order repaired and checked by reading, not compiled, not run on hardware
//
#include <hip/hip_runtime.h>
#include <math.h>

typedef __attribute__((ext_vector_type(16))) __bf16 v16bf;
typedef __attribute__((ext_vector_type(8)))  float v8f;

#define EPSV 1e-8f

__device__ inline unsigned short f2bf(float f) {
  union { float f; unsigned u; } x; x.f = f;
  unsigned u = x.u;
  unsigned r = u + 0x7fffu + ((u >> 16) & 1u);   // round-to-nearest-even
  return (unsigned short)(r >> 16);
}

union BFV { v16bf v; unsigned short u[16]; };

__device__ inline v8f v8zero() {
  v8f r;
#pragma unroll
  for (int i = 0; i < 8; ++i) r[i] = 0.f;
  return r;
}

// ---------------------------------------------------------------------------
// Generic implicit-GEMM convolution with WMMA bf16 (f32 accum).
// K dimension ordered tap-major / channel-minor: k = (ky*KW+kx)*Cin + c.
// All Cin values are multiples of 32, so each 32-wide K chunk sits inside one
// (ky,kx) tap and one tensor of a channel-concat pair -> no div/mod in the
// hot loop, validity + base address computed once per chunk.
// Tile: 64(M) x 64(N), K-chunk 32, 8 waves (wave32), 2 WMMA frags per wave.
// ---------------------------------------------------------------------------
struct ConvP {
  const float* in1; const float* in2;
  const float* w; const float* bias; const float* skip;
  float* out;
  int cin1, cin2;
  int H, W, Cout, Hout, Wout, KH, KW, stride, pad, ldil;
  int M, Ktot, act; // act: 0 none, 1 relu, 2 leaky(0.1)
};

__global__ __launch_bounds__(256) void conv_wmma(ConvP p) {
  __shared__ unsigned short As[64][34];
  __shared__ unsigned short Bs[32][66];
  const int tid  = threadIdx.x;
  const int lane = tid & 31;
  const int wave = tid >> 5;
  const int wm   = wave >> 1;   // 0..3 (M sub-tile)
  const int wn   = wave & 1;    // 0..1 (N sub-tile, 32 cols)
  const int mBase = blockIdx.x * 64;
  const int nBase = blockIdx.y * 64;
  const int KHW = p.KH * p.KW;
  const int HW  = p.Hout * p.Wout;
  const int Cin = p.cin1 + p.cin2;
  const long inHW = (long)p.H * p.W;

  v8f acc0 = v8zero(); v8f acc1 = v8zero();

  // cooperative load coords: 4 threads per row/col, 8 k-elems each
  const int la_row = tid >> 2;
  const int la_k0  = (tid & 3) << 3;
  const int a_m = mBase + la_row;
  const bool a_ok = a_m < p.M;
  int a_n = 0, a_oy = 0, a_ox = 0;
  if (a_ok) { a_n = a_m / HW; int r = a_m - a_n * HW; a_oy = r / p.Wout; a_ox = r - a_oy * p.Wout; }
  const int b_col = nBase + (tid >> 2);
  const bool b_ok = b_col < p.Cout;
  const long wColBase = (long)b_col * p.Ktot;

  // uniform per-chunk state: tap position (ky,kx) and channel block cch
  int ky = 0, kx = 0, cch = 0;

  for (int k0 = 0; k0 < p.Ktot; k0 += 32) {
    // ---- A tile: validity + base computed once per chunk ----
    {
      float v[8];
      const int iyd = a_oy * p.stride - p.pad + ky;
      const int ixd = a_ox * p.stride - p.pad + kx;
      bool ok = a_ok && iyd >= 0 && ixd >= 0 &&
                (iyd % p.ldil) == 0 && (ixd % p.ldil) == 0;
      const int iy = ok ? iyd / p.ldil : 0;
      const int ix = ok ? ixd / p.ldil : 0;
      ok = ok && iy < p.H && ix < p.W;
      const float* src;
      int cloc, cdim;
      if (cch < p.cin1) { src = p.in1; cloc = cch; cdim = p.cin1; }
      else              { src = p.in2; cloc = cch - p.cin1; cdim = p.cin2; }
      const long abase = ((long)(a_n * cdim + cloc + la_k0) * p.H + iy) * p.W + ix;
#pragma unroll
      for (int j = 0; j < 8; ++j)
        v[j] = ok ? src[abase + (long)j * inHW] : 0.f;
#pragma unroll
      for (int j = 0; j < 8; ++j)
        As[la_row][la_k0 + j] = f2bf(v[j]);
    }
    // ---- B tile: weight w[o][c][ky][kx], stride KHW per channel step ----
    {
      const int tap = ky * p.KW + kx;
      const long wbase = wColBase + (long)(cch + la_k0) * KHW + tap;
      float wv[8];
#pragma unroll
      for (int j = 0; j < 8; ++j)
        wv[j] = b_ok ? p.w[wbase + (long)j * KHW] : 0.f;
#pragma unroll
      for (int j = 0; j < 8; ++j)
        Bs[la_k0 + j][tid >> 2] = f2bf(wv[j]);
    }
    __syncthreads();

    // A fragment (16x32 bf16): lanes 0-15 hold K {0..7,16..23}, 16-31 hold {8..15,24..31}
    BFV A;
    const int arow = wm * 16 + (lane & 15);
    const int kh = (lane >> 4) * 8;
#pragma unroll
    for (int i = 0; i < 16; ++i) A.u[i] = As[arow][(i < 8 ? i : i + 8) + kh];
    // B fragments (32x16 bf16): lanes 0-15 K {0..7,16..23} (N=lane), lanes 16-31 K {8..15,24..31}
    {
      const int bcol0 = wn * 32 + (lane & 15);
      const int kb = (lane >> 4) * 16;
      BFV B;
#pragma unroll
      for (int i = 0; i < 16; ++i) B.u[i] = Bs[(i < 8 ? i : i + 8) + kh][bcol0];
      acc0 = __builtin_amdgcn_wmma_f32_16x16x32_bf16(false, A.v, false, B.v, (short)0, acc0, false, false);
      BFV B2;
#pragma unroll
      for (int i = 0; i < 16; ++i) B2.u[i] = Bs[(i < 8 ? i : i + 8) + kh][bcol0 + 16];
      acc1 = __builtin_amdgcn_wmma_f32_16x16x32_bf16(false, A.v, false, B2.v, (short)0, acc1, false, false);
    }
    __syncthreads();

    // advance uniform chunk state: next 32 channels, wrap -> next tap
    cch += 32;
    if (cch == Cin) {
      cch = 0;
      ++kx;
      if (kx == p.KW) { kx = 0; ++ky; }
    }
  }

  // epilogue: C/D layout -> row = vgpr + (lane>>4)*8, col = lane&15
  const int rbase = mBase + wm * 16 + (lane >> 4) * 8;
#pragma unroll
  for (int f = 0; f < 2; ++f) {
    const int col = nBase + wn * 32 + f * 16 + (lane & 15);
    const float bv = p.bias ? p.bias[col] : 0.f;
#pragma unroll
    for (int r = 0; r < 8; ++r) {
      const int m = rbase + r;
      if (m >= p.M || col >= p.Cout) continue;
      int n = m / HW; int rem = m - n * HW;
      int oy = rem / p.Wout; int ox = rem - oy * p.Wout;
      long oidx = ((long)(n * p.Cout + col) * p.Hout + oy) * p.Wout + ox;
      float v = (f == 0 ? acc0[r] : acc1[r]) + bv;
      if (p.skip) v += p.skip[oidx];
      if (p.act == 1) v = v > 0.f ? v : 0.f;
      else if (p.act == 2) v = v > 0.f ? v : 0.1f * v;
      p.out[oidx] = v;
    }
  }
}

// ---------------------------------------------------------------------------
// Vector quantize: per 16-row tile, dist' = |cb|^2 - 2 z.cb over all 2048
// codes via WMMA (16 waves x 128 codes), then argmin + softmax(-dist) + zq
// gather + VQ loss. (|z|^2 cancels in argmin and row-softmax.)
// ---------------------------------------------------------------------------
struct VQP {
  const float* z; const float* cb; const float* cbn;
  float* zq; float* probs; float* accLoss;
  int C, H, W, M, Kc;
};

__global__ __launch_bounds__(512) void vq_wmma(VQP p) {
  __shared__ float s_minv[16][16];
  __shared__ int   s_mini[16][16];
  __shared__ float s_sum[16][16];
  __shared__ float rowMin[16];
  __shared__ int   rowIdx[16];
  __shared__ float rowZ[16];
  __shared__ float red[512];

  const int tid = threadIdx.x;
  const int lane = tid & 31;
  const int wave = tid >> 5;           // 0..15, each owns 128 codes
  const int mBase = blockIdx.x * 16;
  const int HW = p.H * p.W;

  v8f acc[8];
#pragma unroll
  for (int f = 0; f < 8; ++f) acc[f] = v8zero();

  const int am = mBase + (lane & 15);
  const int an = am / HW; const int arem = am - an * HW;
  const int ay = arem / p.W; const int ax = arem - ay * p.W;
  const int kh = (lane >> 4) * 8;
  const int kb = (lane >> 4) * 16;
  const long zbase = ((long)an * p.C * p.H + ay) * 1 + 0; // recomputed below

  for (int k0 = 0; k0 < p.C; k0 += 32) {
    BFV A;
#pragma unroll
    for (int i = 0; i < 16; ++i) {
      int c = k0 + (i < 8 ? i : i + 8) + kh;
      A.u[i] = f2bf(p.z[((long)(an * p.C + c) * p.H + ay) * p.W + ax]);
    }
#pragma unroll
    for (int f = 0; f < 8; ++f) {
      const int code = wave * 128 + f * 16 + (lane & 15);
      BFV B;
#pragma unroll
      for (int i = 0; i < 16; ++i)
        B.u[i] = f2bf(p.cb[(long)code * p.C + k0 + (i < 8 ? i : i + 8) + kh]);
      acc[f] = __builtin_amdgcn_wmma_f32_16x16x32_bf16(false, A.v, false, B.v, (short)0, acc[f], false, false);
    }
  }

  float cn[8];
#pragma unroll
  for (int f = 0; f < 8; ++f) cn[f] = p.cbn[wave * 128 + f * 16 + (lane & 15)];

  // per-row argmin within wave (16-lane halves via shfl_xor masks < 16)
#pragma unroll
  for (int r = 0; r < 8; ++r) {
    const int row = (lane >> 4) * 8 + r;
    float mv = 3.4e38f; int mi = 0;
#pragma unroll
    for (int f = 0; f < 8; ++f) {
      float d = cn[f] - 2.f * acc[f][r];
      int code = wave * 128 + f * 16 + (lane & 15);
      if (d < mv || (d == mv && code < mi)) { mv = d; mi = code; }
    }
    for (int msk = 1; msk < 16; msk <<= 1) {
      float ov = __shfl_xor(mv, msk, 32);
      int   oi = __shfl_xor(mi, msk, 32);
      if (ov < mv || (ov == mv && oi < mi)) { mv = ov; mi = oi; }
    }
    if ((lane & 15) == 0) { s_minv[row][wave] = mv; s_mini[row][wave] = mi; }
  }
  __syncthreads();
  if (tid < 16) {
    float mv = s_minv[tid][0]; int mi = s_mini[tid][0];
    for (int w2 = 1; w2 < 16; ++w2) {
      float ov = s_minv[tid][w2]; int oi = s_mini[tid][w2];
      if (ov < mv || (ov == mv && oi < mi)) { mv = ov; mi = oi; }
    }
    rowMin[tid] = mv; rowIdx[tid] = mi;
  }
  __syncthreads();

  // softmax denominator
#pragma unroll
  for (int r = 0; r < 8; ++r) {
    const int row = (lane >> 4) * 8 + r;
    const float rm = rowMin[row];
    float s = 0.f;
#pragma unroll
    for (int f = 0; f < 8; ++f) {
      float d = cn[f] - 2.f * acc[f][r];
      s += __expf(rm - d);
    }
    for (int msk = 1; msk < 16; msk <<= 1) s += __shfl_xor(s, msk, 32);
    if ((lane & 15) == 0) s_sum[row][wave] = s;
  }
  __syncthreads();
  if (tid < 16) {
    float s = 0.f;
    for (int w2 = 0; w2 < 16; ++w2) s += s_sum[tid][w2];
    rowZ[tid] = s;
  }
  __syncthreads();

  // write probs
#pragma unroll
  for (int r = 0; r < 8; ++r) {
    const int row = (lane >> 4) * 8 + r;
    const float rm = rowMin[row];
    const float invZ = 1.f / rowZ[row];
    const int m = mBase + row;
    if (m >= p.M) continue;
#pragma unroll
    for (int f = 0; f < 8; ++f) {
      int code = wave * 128 + f * 16 + (lane & 15);
      float d = cn[f] - 2.f * acc[f][r];
      p.probs[(long)m * p.Kc + code] = __expf(rm - d) * invZ;
    }
  }

  // zq gather (NCHW) + VQ loss partial
  {
    const int row = tid >> 5;              // 16 rows / 512 threads
    const int per = p.C / 32;              // 16 channels per thread
    const int c0 = (tid & 31) * per;
    const int m = mBase + row;
    const int n = m / HW; const int rem = m - n * HW;
    const int y = rem / p.W; const int x = rem - y * p.W;
    const int idx = rowIdx[row];
    float part = 0.f;
    for (int c = c0; c < c0 + per; ++c) {
      float zv = p.cb[(long)idx * p.C + c];
      long zidx = ((long)(n * p.C + c) * p.H + y) * p.W + x;
      float zf = p.z[zidx];
      p.zq[zidx] = zv;
      float dlt = zv - zf;
      part += dlt * dlt;
    }
    red[tid] = part;
  }
  __syncthreads();
  for (int s2 = 256; s2 > 0; s2 >>= 1) {
    if (tid < s2) red[tid] += red[tid + s2];
    __syncthreads();
  }
  if (tid == 0) atomicAdd(p.accLoss, red[0]);
}

// ---------------------------------------------------------------------------
// Small helper kernels
// ---------------------------------------------------------------------------
__global__ void cbnorm_kernel(const float* cb, float* cbn, int K, int C) {
  int i = blockIdx.x * blockDim.x + threadIdx.x;
  if (i < K) {
    float s = 0.f;
    const float* r = cb + (long)i * C;
    for (int c = 0; c < C; ++c) s += r[c] * r[c];
    cbn[i] = s;
  }
}

__global__ __launch_bounds__(256) void jsd_kernel(const float* probs, int half, int Kc,
                                                  float scale, float* acc) {
  __shared__ float red[256];
  const int row = blockIdx.x;
  const float* pr = probs + (long)row * Kc;
  const float* qr = probs + (long)(row + half) * Kc;
  float s = 0.f;
  for (int c = threadIdx.x; c < Kc; c += 256) {
    float a = pr[c], b = qr[c];
    float m = 0.5f * (a + b);
    float lm = logf(m + EPSV);
    s += a * (logf(a + EPSV) - lm) + b * (logf(b + EPSV) - lm);
  }
  red[threadIdx.x] = s;
  __syncthreads();
  for (int s2 = 128; s2 > 0; s2 >>= 1) {
    if (threadIdx.x < s2) red[threadIdx.x] += red[threadIdx.x + s2];
    __syncthreads();
  }
  if (threadIdx.x == 0) atomicAdd(acc, scale * red[0]);
}

__global__ __launch_bounds__(256) void sqdiff_kernel(const float* a, const float* b,
                                                     long n, float* acc) {
  __shared__ float red[256];
  float s = 0.f;
  for (long i = (long)blockIdx.x * 256 + threadIdx.x; i < n; i += (long)gridDim.x * 256) {
    float d = a[i] - b[i];
    s += d * d;
  }
  red[threadIdx.x] = s;
  __syncthreads();
  for (int s2 = 128; s2 > 0; s2 >>= 1) {
    if (threadIdx.x < s2) red[threadIdx.x] += red[threadIdx.x + s2];
    __syncthreads();
  }
  if (threadIdx.x == 0) atomicAdd(acc, red[0]);
}

__global__ void zero_kernel(float* p, int n) {
  int i = blockIdx.x * blockDim.x + threadIdx.x;
  if (i < n) p[i] = 0.f;
}

__global__ void copy_kernel(const float* src, float* dst, long n) {
  for (long i = (long)blockIdx.x * 256 + threadIdx.x; i < n; i += (long)gridDim.x * 256)
    dst[i] = src[i];
}

__global__ void finalize_kernel(const float* acc, float* outLoss) {
  if (threadIdx.x == 0) {
    outLoss[0] = acc[0] / 4816896.f;                                    // recon mean
    outLoss[1] = 1.25f * (acc[1] / (3136.f * 512.f) +                   // (1+beta)*mse
                          acc[2] / (12544.f * 512.f));
    outLoss[2] = acc[3];                                                // contra
  }
}

// ---------------------------------------------------------------------------
// Host side
// ---------------------------------------------------------------------------
static void launch_conv(hipStream_t st, const float* in1, const float* in2, int cin1, int cin2,
                        const float* w, const float* bias, const float* skip, float* out,
                        int N, int H, int W, int Cout, int KH, int KW,
                        int stride, int pad, int ldil, int act) {
  ConvP p;
  p.in1 = in1; p.in2 = in2; p.w = w; p.bias = bias; p.skip = skip; p.out = out;
  p.cin1 = cin1; p.cin2 = cin2;
  p.H = H; p.W = W; p.Cout = Cout; p.KH = KH; p.KW = KW;
  p.stride = stride; p.pad = pad; p.ldil = ldil; p.act = act;
  const int Hd = H * ldil - (ldil - 1);
  p.Hout = (Hd + 2 * pad - KH) / stride + 1;
  p.Wout = (W * ldil - (ldil - 1) + 2 * pad - KW) / stride + 1;
  p.M = N * p.Hout * p.Wout;
  p.Ktot = (cin1 + cin2) * KH * KW;
  dim3 grid((p.M + 63) / 64, (Cout + 63) / 64);
  conv_wmma<<<grid, 256, 0, st>>>(p);
}

static void launch_vq(hipStream_t st, const float* z, const float* cb, const float* cbn,
                      float* zq, float* probs, float* accLoss, int C, int H, int W, int N) {
  VQP p;
  p.z = z; p.cb = cb; p.cbn = cbn; p.zq = zq; p.probs = probs; p.accLoss = accLoss;
  p.C = C; p.H = H; p.W = W; p.M = N * H * W; p.Kc = 2048;
  vq_wmma<<<p.M / 16, 512, 0, st>>>(p);
}

extern "C" void kernel_launch(void* const* d_in, const int* in_sizes, int n_in,
                              void* d_out, int out_size, void* d_ws, size_t ws_size,
                              hipStream_t stream) {
  const int NB = 16, FEAT = 384, HID = 512, E = 512, Kc = 2048;
  const int H = 28, W = 28, H2 = 14, W2 = 14;

  const float* x = (const float*)d_in[0];
  auto P = [&](int i) { return (const float*)d_in[i]; };
  const float* enc0_w1 = P(1);  const float* enc0_b1 = P(2);
  const float* enc0_w2 = P(3);  const float* enc0_b2 = P(4);
  const float* enc0_ws = P(5);  const float* enc0_bs = P(6);
  const float* enc1_w1 = P(7);  const float* enc1_b1 = P(8);
  const float* enc1_w2 = P(9);  const float* enc1_b2 = P(10);
  const float* top_w1  = P(11); const float* top_b1  = P(12);
  const float* top_w2  = P(13); const float* top_b2  = P(14);
  const float* in0_w   = P(15);
  const float* cb0     = P(16); const float* cb1     = P(17);
  const float* up_w    = P(18); const float* up_b    = P(19);
  const float* outp1_w = P(20); const float* outp1_b = P(21);
  const float* agg_w   = P(22); const float* agg_b   = P(23);
  const float* dec0_w1 = P(24); const float* dec0_b1 = P(25);
  const float* dec0_w2 = P(26); const float* dec0_b2 = P(27);
  const float* dec1_w1 = P(28); const float* dec1_b1 = P(29);
  const float* dec1_w2 = P(30); const float* dec1_b2 = P(31);
  const float* dec1_ws = P(32); const float* dec1_bs = P(33);

  // ---- workspace layout (floats) ----
  const size_t BIG = (size_t)NB * HID * H * W;     // 6,422,528
  const size_t SML = (size_t)NB * E * H2 * W2;     // 1,605,632
  const size_t SB1s = (size_t)NB * (HID / 2) * H2 * W2;
  const size_t PB0s = (size_t)NB * H2 * W2 * Kc;   // 6,422,528
  const size_t PB1s = (size_t)NB * H * W * Kc;     // 25,690,112
  const size_t RBs  = (size_t)NB * FEAT * H * W;   // 4,816,896

  float* ws = (float*)d_ws;
  size_t off = 0;
  float* acc  = ws + off; off += 16;   // [0]=recon [1]=vq0 [2]=vq1 [3]=contra
  float* cb0n = ws + off; off += 2048;
  float* cb1n = ws + off; off += 2048;
  float* BUF1 = ws + off; off += BIG;
  float* BUF2 = ws + off; off += BIG;
  float* BUF3 = ws + off; off += BIG;
  float* SB1  = ws + off; off += SB1s;
  float* SB2  = ws + off; off += SML;
  float* SB3  = ws + off; off += SML;
  float* SB4  = ws + off; off += SML;
  float* PB0  = ws + off; off += PB0s;
  float* PB1  = ws + off; off += PB1s;
  float* RB   = ws + off; off += RBs;
  (void)ws_size; (void)n_in; (void)in_sizes; (void)out_size;

  zero_kernel<<<1, 16, 0, stream>>>(acc, 16);
  cbnorm_kernel<<<8, 256, 0, stream>>>(cb0, cb0n, Kc, E);
  cbnorm_kernel<<<8, 256, 0, stream>>>(cb1, cb1n, Kc, E);

  // enc0 res-block: skip=1x1(x), h=relu(3x3(x)), fb=relu(3x3(h)+skip)
  launch_conv(stream, x, nullptr, FEAT, 0, enc0_ws, enc0_bs, nullptr, BUF3, NB, H, W, HID, 1, 1, 1, 0, 1, 0);
  launch_conv(stream, x, nullptr, FEAT, 0, enc0_w1, enc0_b1, nullptr, BUF2, NB, H, W, HID, 3, 3, 1, 1, 1, 1);
  launch_conv(stream, BUF2, nullptr, HID, 0, enc0_w2, enc0_b2, BUF3, BUF1, NB, H, W, HID, 3, 3, 1, 1, 1, 1);
  // enc1 res-block (identity skip, in-place is element-safe)
  launch_conv(stream, BUF1, nullptr, HID, 0, enc1_w1, enc1_b1, nullptr, BUF2, NB, H, W, HID, 3, 3, 1, 1, 1, 1);
  launch_conv(stream, BUF2, nullptr, HID, 0, enc1_w2, enc1_b2, BUF1, BUF1, NB, H, W, HID, 3, 3, 1, 1, 1, 1);
  // top: 4x4 s2 relu -> 3x3 (+leaky fused, feeds in0 only)
  launch_conv(stream, BUF1, nullptr, HID, 0, top_w1, top_b1, nullptr, SB1, NB, H, W, HID / 2, 4, 4, 2, 1, 1, 1);
  launch_conv(stream, SB1, nullptr, HID / 2, 0, top_w2, top_b2, nullptr, SB2, NB, H2, W2, HID, 3, 3, 1, 1, 1, 2);
  // f0 = 1x1(leaky(ft)), no bias
  launch_conv(stream, SB2, nullptr, HID, 0, in0_w, nullptr, nullptr, SB3, NB, H2, W2, E, 1, 1, 1, 0, 1, 0);
  // VQ0
  launch_vq(stream, SB3, cb0, cb0n, SB4, PB0, acc + 1, E, H2, W2, NB);
  // up = transposed conv (lhs_dilation=2, pad=2, 4x4) : 14x14 -> 28x28
  launch_conv(stream, SB4, nullptr, E, 0, up_w, up_b, nullptr, BUF2, NB, H2, W2, E, 4, 4, 1, 2, 2, 0);
  // f1 = 1x1(concat(fb, up))
  launch_conv(stream, BUF1, BUF2, HID, E, outp1_w, outp1_b, nullptr, BUF3, NB, H, W, HID, 1, 1, 1, 0, 1, 0);
  // VQ1 (zq1 overwrites fb, which is now dead)
  launch_vq(stream, BUF3, cb1, cb1n, BUF1, PB1, acc + 2, E, H, W, NB);
  // feat = 1x1(concat(up, zq1))
  launch_conv(stream, BUF2, BUF1, E, E, agg_w, agg_b, nullptr, BUF3, NB, H, W, HID, 1, 1, 1, 0, 1, 0);
  // dec0 res-block (identity skip = feat)
  launch_conv(stream, BUF3, nullptr, HID, 0, dec0_w1, dec0_b1, nullptr, BUF2, NB, H, W, HID, 3, 3, 1, 1, 1, 1);
  launch_conv(stream, BUF2, nullptr, HID, 0, dec0_w2, dec0_b2, BUF3, BUF1, NB, H, W, HID, 3, 3, 1, 1, 1, 1);
  // dec1 res-block (1x1 skip to 384ch)
  launch_conv(stream, BUF1, nullptr, HID, 0, dec1_w1, dec1_b1, nullptr, BUF2, NB, H, W, HID, 3, 3, 1, 1, 1, 1);
  launch_conv(stream, BUF1, nullptr, HID, 0, dec1_ws, dec1_bs, nullptr, RB, NB, H, W, FEAT, 1, 1, 1, 0, 1, 0);
  launch_conv(stream, BUF2, nullptr, HID, 0, dec1_w2, dec1_b2, RB, RB, NB, H, W, FEAT, 3, 3, 1, 1, 1, 1);

  // losses
  sqdiff_kernel<<<2048, 256, 0, stream>>>(RB, x, (long)RBs, acc + 0);
  const int half0 = NB * H2 * W2 / 2;   // 1568
  const int half1 = NB * H * W / 2;     // 6272
  jsd_kernel<<<half0, 256, 0, stream>>>(PB0, half0, Kc, 0.5f / (float)half0, acc + 3);
  jsd_kernel<<<half1, 256, 0, stream>>>(PB1, half1, Kc, -0.01f * 0.5f / (float)half1, acc + 3);

  // outputs: feat[:8] (contiguous first half of NCHW buffer) then 3 losses
  const long featHalf = (long)(NB / 2) * HID * H * W;  // 3,211,264
  copy_kernel<<<2048, 256, 0, stream>>>(BUF3, (float*)d_out, featHalf);
  finalize_kernel<<<1, 32, 0, stream>>>(acc, (float*)d_out + featHalf);
}
